// MoeRouter_48215302865690
// MI455X (gfx1250) — matrix operand order repaired and checked by reading, not compiled, not run on hardware
//
#include <hip/hip_runtime.h>
#include <hip/hip_bf16.h>

typedef __attribute__((ext_vector_type(16))) __bf16 v16bf;
typedef __attribute__((ext_vector_type(8)))  float  v8f;
typedef __attribute__((ext_vector_type(4)))  float  v4f;

#define N_TOK 32768
#define DDIM  1024
#define NEXP  64
#define ROWS_PER_BLOCK 128
#define THREADS 256
#define LDS_STRIDE 65   // 64 experts + 1 pad to avoid 64-bank conflicts
#define NBLOCKS (N_TOK / ROWS_PER_BLOCK)
#define KSTEPS (DDIM / 32)

// ---------------- Kernel 0: gate_w fp32 -> bf16 scratch -------------------
__global__ void cvt_gate_kernel(const float* __restrict__ gw,
                                __bf16* __restrict__ gb) {
    int i = blockIdx.x * blockDim.x + threadIdx.x;
    if (i < NEXP * DDIM) gb[i] = (__bf16)gw[i];
}

// ---------------- Kernel 1: WMMA logits + softmax + top2 ------------------
__global__ __launch_bounds__(THREADS)
void moe_router_kernel(const float* __restrict__ x,
                       const __bf16* __restrict__ gb,
                       float* __restrict__ out_w,
                       float* __restrict__ out_i,
                       float* __restrict__ partial) {
    __shared__ float lds[ROWS_PER_BLOCK * LDS_STRIDE];

    const int tid    = threadIdx.x;
    const int lane   = tid & 31;
    const int wave   = tid >> 5;
    const int mlocal = lane & 15;   // row within the wave's 16-row tile
    const int hi     = lane >> 4;   // half-wave select (K striping)

    const int row0 = blockIdx.x * ROWS_PER_BLOCK + wave * 16;

    v8f acc[4] = {};                // 4 tiles of 16 experts each

    const float* xrow = x + (size_t)(row0 + mlocal) * DDIM;
    const __bf16* grow = gb + (size_t)mlocal * DDIM + hi * 16;

    // ---- software-pipelined A stream: preload K-chunk 0 ----------------
    // A k order used here (contiguous, B matches): lanes 0-15: row=lane, elems 0..15 = K+0..15;
    // lanes 16-31: row=lane-16, elems 0..15 = K+16..31.
    v4f f0, f1, f2, f3;
    {
        const int ka = hi * 16;
        f0 = *(const v4f*)(xrow + ka);
        f1 = *(const v4f*)(xrow + ka + 4);
        f2 = *(const v4f*)(xrow + ka + 8);
        f3 = *(const v4f*)(xrow + ka + 12);
    }

    for (int ks = 0; ks < KSTEPS; ++ks) {
        const int k0 = ks * 32;

        // ---- convert staged fp32 A chunk -> bf16 WMMA operand ----------
        v16bf a;
        #pragma unroll
        for (int i = 0; i < 4; ++i) {
            a[i]      = (__bf16)f0[i];
            a[4 + i]  = (__bf16)f1[i];
            a[8 + i]  = (__bf16)f2[i];
            a[12 + i] = (__bf16)f3[i];
        }

        // ---- issue ALL B-tile loads first (8x b128, L2-resident) -------
        // B k order used here (contiguous, as A): lane = expert column, half-wave = K+16.
        v16bf bt[4];
        #pragma unroll
        for (int t = 0; t < 4; ++t) {
            bt[t] = *(const v16bf*)(grow + (size_t)t * 16 * DDIM + k0);
        }

        // ---- issue next A chunk loads to overlap with WMMAs ------------
        if (ks + 1 < KSTEPS) {
            const int ka = k0 + 32 + hi * 16;
            f0 = *(const v4f*)(xrow + ka);
            f1 = *(const v4f*)(xrow + ka + 4);
            f2 = *(const v4f*)(xrow + ka + 8);
            f3 = *(const v4f*)(xrow + ka + 12);
            __builtin_prefetch(xrow + k0 + 64, 0, 1);   // global_prefetch_b8
        }

        // ---- 4 back-to-back WMMAs with staggered load waits ------------
        #pragma unroll
        for (int t = 0; t < 4; ++t) {
            acc[t] = __builtin_amdgcn_wmma_f32_16x16x32_bf16(
                false, a, false, bt[t], (short)0, acc[t], false, false);
        }
    }

    // ---- scatter logits to LDS (C layout: lanes0-15 M=r, lanes16-31 M=r+8)
    #pragma unroll
    for (int t = 0; t < 4; ++t) {
        #pragma unroll
        for (int r = 0; r < 8; ++r) {
            int rl = wave * 16 + hi * 8 + r;
            lds[rl * LDS_STRIDE + t * 16 + mlocal] = acc[t][r];
        }
    }
    __syncthreads();

    // ---- phase 2: one thread per row: softmax + top2 + renorm ----------
    if (tid < ROWS_PER_BLOCK) {
        float* L = lds + tid * LDS_STRIDE;
        float mx = L[0];
        #pragma unroll 8
        for (int e = 1; e < NEXP; ++e) mx = fmaxf(mx, L[e]);
        float s = 0.f;
        #pragma unroll 8
        for (int e = 0; e < NEXP; ++e) { float p = __expf(L[e] - mx); L[e] = p; s += p; }
        float inv = 1.f / s;
        float b0 = -1.f, b1 = -1.f; int i0 = 0, i1 = 0;
        for (int e = 0; e < NEXP; ++e) {
            float p = L[e] * inv;
            L[e] = p;                                // store prob for phase 3
            if (p > b0)      { b1 = b0; i1 = i0; b0 = p; i0 = e; }
            else if (p > b1) { b1 = p; i1 = e; }
        }
        float rs = 1.f / (b0 + b1 + 1e-9f);
        int grow_idx = blockIdx.x * ROWS_PER_BLOCK + tid;
        out_w[grow_idx * 2 + 0] = b0 * rs;
        out_w[grow_idx * 2 + 1] = b1 * rs;
        out_i[grow_idx * 2 + 0] = (float)i0;
        out_i[grow_idx * 2 + 1] = (float)i1;
    }
    __syncthreads();

    // ---- phase 3: deterministic per-expert partial sums ----------------
    if (tid < NEXP) {
        float s = 0.f;
        for (int r = 0; r < ROWS_PER_BLOCK; ++r) s += lds[r * LDS_STRIDE + tid];
        partial[blockIdx.x * NEXP + tid] = s;
    }
}

// ---------------- Kernel 2: aux loss reduction (deterministic) ------------
__global__ void aux_loss_kernel(const float* __restrict__ partial,
                                float* __restrict__ out) {
    __shared__ float terms[NEXP];
    int e = threadIdx.x;
    float s = 0.f;
    for (int b = 0; b < NBLOCKS; ++b) s += partial[b * NEXP + e];
    float m = s * (1.0f / (float)N_TOK);
    terms[e] = m * __logf(m + 1e-9f);
    __syncthreads();
    if (e == 0) {
        float t = 0.f;
        for (int i = 0; i < NEXP; ++i) t += terms[i];
        out[0] = t;
    }
}

// --------------------------------------------------------------------------
extern "C" void kernel_launch(void* const* d_in, const int* in_sizes, int n_in,
                              void* d_out, int out_size, void* d_ws, size_t ws_size,
                              hipStream_t stream) {
    const float* x  = (const float*)d_in[0];   // [32768, 1024] fp32
    const float* gw = (const float*)d_in[1];   // [64, 1024] fp32

    float* out   = (float*)d_out;
    float* out_w = out;                        // [N,2] top-k weights
    float* out_i = out + (size_t)N_TOK * 2;    // [N,2] indices (as float)
    float* out_a = out + (size_t)N_TOK * 4;    // aux loss scalar

    // workspace: [0, 128KB) gate bf16, then per-block partial prob sums
    __bf16* gb      = (__bf16*)d_ws;
    float*  partial = (float*)((char*)d_ws + (size_t)NEXP * DDIM * sizeof(__bf16));

    cvt_gate_kernel<<<(NEXP * DDIM + 255) / 256, 256, 0, stream>>>(gw, gb);
    moe_router_kernel<<<NBLOCKS, THREADS, 0, stream>>>(x, gb, out_w, out_i, partial);
    aux_loss_kernel<<<1, NEXP, 0, stream>>>(partial, out_a);
}
